// TreeDualLSTM_47502338294510
// MI455X (gfx1250) — matrix operand order repaired and checked by reading, not compiled, not run on hardware
//
#include <hip/hip_runtime.h>
#include <hip/hip_bf16.h>

typedef __bf16 bf16_t;
typedef __attribute__((ext_vector_type(16))) __bf16 v16bf;
typedef __attribute__((ext_vector_type(8)))  float  v8f;

#define BATCH  128
#define NNODES 511
#define MDIM   256
#define RDIM   64
#define INDIM  512

__device__ __forceinline__ float sigf(float x) { return 1.0f / (1.0f + __expf(-x)); }

// ---------------------------------------------------------------------------
// Tiled bf16 GEMM with f32 accumulate via v_wmma_f32_16x16x32_bf16.
//   C[r, n] = sum_k A[r, k] * Bt[n, k]  (+ bias[n])  (+ Cin[r, n])
// A: row-major bf16, leading dim lda (lda >= K, multiple of 16)
// Bt: column-major weights (i.e. Bt[n * ldb + k]), bf16
// Block = 128 threads (4 waves), computes a 128x64 tile of C.
// Each wave owns a 32x64 slice: 2 A-frags x 4 B-frags = 8 WMMAs / K-step,
// so each B fragment fetched from LDS is reused by 2 WMMAs.
// rows must be a multiple of 128; cols multiple of 64; K multiple of 32.
// ---------------------------------------------------------------------------
__global__ __launch_bounds__(128)
void gemm_bf16_wmma(const bf16_t* __restrict__ A, int lda,
                    const bf16_t* __restrict__ Bt, int ldb,
                    const float* __restrict__ bias,
                    const float* Cin,
                    float* C, int ldc, int K)
{
    __shared__ unsigned int Asm_[128][17];  // 128 rows x 16 dwords (32 bf16) + pad
    __shared__ unsigned int Bsm_[64][17];   //  64 cols x 16 dwords (32 bf16) + pad

    const int tid   = threadIdx.x;
    const int wave  = tid >> 5;
    const int lane  = tid & 31;
    const int row0  = blockIdx.y * 128;
    const int col0  = blockIdx.x * 64;

    v8f acc[2][4];
#pragma unroll
    for (int mt = 0; mt < 2; ++mt)
#pragma unroll
        for (int nt = 0; nt < 4; ++nt)
            acc[mt][nt] = (v8f){0.f,0.f,0.f,0.f,0.f,0.f,0.f,0.f};

    const int hi  = lane >> 4;
    const int nlo = lane & 15;
    const int bcl = tid >> 1;   // B staging: col 0..63
    const int bhf = tid & 1;    // which 16-element half

    for (int k0 = 0; k0 < K; k0 += 32) {
        // ---- stage A tile: 128 rows x 32 bf16, one full row per thread ----
        {
            const uint4* src = (const uint4*)(A + (size_t)(row0 + tid) * lda + k0);
            uint4 d0 = src[0], d1 = src[1], d2 = src[2], d3 = src[3];
            Asm_[tid][0]  = d0.x; Asm_[tid][1]  = d0.y; Asm_[tid][2]  = d0.z; Asm_[tid][3]  = d0.w;
            Asm_[tid][4]  = d1.x; Asm_[tid][5]  = d1.y; Asm_[tid][6]  = d1.z; Asm_[tid][7]  = d1.w;
            Asm_[tid][8]  = d2.x; Asm_[tid][9]  = d2.y; Asm_[tid][10] = d2.z; Asm_[tid][11] = d2.w;
            Asm_[tid][12] = d3.x; Asm_[tid][13] = d3.y; Asm_[tid][14] = d3.z; Asm_[tid][15] = d3.w;
        }
        // ---- stage B tile: 64 cols x 32 bf16 (column-major weights) ----
        {
            const uint4* src = (const uint4*)(Bt + (size_t)(col0 + bcl) * ldb + k0 + bhf * 16);
            uint4 d0 = src[0], d1 = src[1];
            int c = bhf * 8;
            Bsm_[bcl][c + 0] = d0.x; Bsm_[bcl][c + 1] = d0.y;
            Bsm_[bcl][c + 2] = d0.z; Bsm_[bcl][c + 3] = d0.w;
            Bsm_[bcl][c + 4] = d1.x; Bsm_[bcl][c + 5] = d1.y;
            Bsm_[bcl][c + 6] = d1.z; Bsm_[bcl][c + 7] = d1.w;
        }
        __syncthreads();

        // A fragments: 16x32 bf16.  Lane (row = lane&15, hi = lane>>4):
        //   hi=0 -> K dwords {0..3, 8..11}, hi=1 -> {4..7, 12..15}
        union { unsigned int u[8]; v16bf v; } afrag[2];
#pragma unroll
        for (int mt = 0; mt < 2; ++mt) {
            int arow = wave * 32 + mt * 16 + nlo;
#pragma unroll
            for (int d = 0; d < 4; ++d) {
                afrag[mt].u[d]     = Asm_[arow][hi * 4 + d];
                afrag[mt].u[4 + d] = Asm_[arow][8 + hi * 4 + d];
            }
        }

#pragma unroll
        for (int nt = 0; nt < 4; ++nt) {
            // B fragment: 32x16 bf16.  Lane (col = lane&15, hi): K dwords {hi*4 + 0..3, 8 + hi*4 + 0..3}
            int bcol = nt * 16 + nlo;
            union { unsigned int u[8]; v16bf v; } bfrag;
#pragma unroll
            for (int d = 0; d < 4; ++d) { bfrag.u[d] = Bsm_[bcol][hi * 4 + d]; bfrag.u[4 + d] = Bsm_[bcol][8 + hi * 4 + d]; }
#pragma unroll
            for (int mt = 0; mt < 2; ++mt) {
                acc[mt][nt] = __builtin_amdgcn_wmma_f32_16x16x32_bf16(
                    false, afrag[mt].v, false, bfrag.v, (short)0, acc[mt][nt], false, false);
            }
        }
        __syncthreads();
    }

    // C layout: lane (hi, nlo), VGPR v -> (M = v + 8*hi, N = nlo)
#pragma unroll
    for (int nt = 0; nt < 4; ++nt) {
        int col = col0 + nt * 16 + nlo;
        float badd = bias ? bias[col] : 0.0f;
#pragma unroll
        for (int mt = 0; mt < 2; ++mt) {
#pragma unroll
            for (int v = 0; v < 8; ++v) {
                int rr = row0 + wave * 32 + mt * 16 + v + 8 * hi;
                size_t idx = (size_t)rr * ldc + col;
                float val = acc[mt][nt][v] + badd;
                if (Cin) val += Cin[idx];
                C[idx] = val;
            }
        }
    }
}

// ---------------------------------------------------------------------------
// Packing / elementwise kernels
// ---------------------------------------------------------------------------

// A_seq row r=(p,b): [ h_child(256) | rel_child(64) ] -> bf16, lda = 320
__global__ void pack_child(bf16_t* __restrict__ A, const float* __restrict__ hbuf,
                           const float* __restrict__ rel, int base, int childoff, int rows)
{
    long long idx = (long long)blockIdx.x * blockDim.x + threadIdx.x;
    if (idx >= (long long)rows * 320) return;
    int r = (int)(idx / 320), k = (int)(idx % 320);
    int p = r >> 7, b = r & 127;
    int child = 2 * (base + p) + childoff;
    float v;
    if (k < MDIM) v = hbuf[((size_t)child * BATCH + b) * MDIM + k];
    else          v = rel[((size_t)b * NNODES + child) * RDIM + (k - MDIM)];
    A[(size_t)r * 320 + k] = (bf16_t)v;
}

// X row r=(p,b): word_tag_emb[b, base+p, :] -> bf16, lda = 512
__global__ void pack_x(bf16_t* __restrict__ X, const float* __restrict__ wte,
                       int base, int rows)
{
    long long idx = (long long)blockIdx.x * blockDim.x + threadIdx.x;
    if (idx >= (long long)rows * INDIM) return;
    int r = (int)(idx / INDIM), k = (int)(idx % INDIM);
    int p = r >> 7, b = r & 127;
    X[idx] = (bf16_t)wte[((size_t)b * NNODES + (base + p)) * INDIM + k];
}

// Seq-LSTM sub-cell, child 0 (prev h=c=0): c = sig(i)*tanh(g); h = sig(o)*tanh(c)
__global__ void seq_cell0(const float* __restrict__ g, float* __restrict__ hs,
                          float* __restrict__ cs, bf16_t* __restrict__ hsb, int rows)
{
    long long idx = (long long)blockIdx.x * blockDim.x + threadIdx.x;
    if (idx >= (long long)rows * MDIM) return;
    int r = (int)(idx / MDIM), m = (int)(idx % MDIM);
    const float* gr = g + (size_t)r * 1024;
    float c = sigf(gr[m]) * tanhf(gr[512 + m]);
    float h = sigf(gr[768 + m]) * tanhf(c);
    cs[idx] = c; hs[idx] = h; hsb[idx] = (bf16_t)h;
}

// Seq-LSTM sub-cell, child 1: c2 = sig(f)*c + sig(i)*tanh(g); h_tilde = sig(o)*tanh(c2)
__global__ void seq_cell1(const float* __restrict__ g, float* __restrict__ hs,
                          const float* __restrict__ cs, bf16_t* __restrict__ hsb, int rows)
{
    long long idx = (long long)blockIdx.x * blockDim.x + threadIdx.x;
    if (idx >= (long long)rows * MDIM) return;
    int r = (int)(idx / MDIM), m = (int)(idx % MDIM);
    const float* gr = g + (size_t)r * 1024;
    float c = sigf(gr[256 + m]) * cs[idx] + sigf(gr[m]) * tanhf(gr[512 + m]);
    float h = sigf(gr[768 + m]) * tanhf(c);
    hs[idx] = h; hsb[idx] = (bf16_t)h;
}

// Tree cell, internal node.  gx cols: [i|f|u|o]*256, gh cols: [ih|uh|oh]*256
__global__ void final_gates(const float* __restrict__ gx, const float* __restrict__ gh,
                            const float* __restrict__ f0p, const float* __restrict__ f1p,
                            float* __restrict__ hbuf, float* __restrict__ cbuf,
                            int base, int rows)
{
    long long idx = (long long)blockIdx.x * blockDim.x + threadIdx.x;
    if (idx >= (long long)rows * MDIM) return;
    int r = (int)(idx / MDIM), m = (int)(idx % MDIM);
    int p = r >> 7, b = r & 127;
    int node = base + p;
    const float* gxr = gx + (size_t)r * 1024;
    const float* ghr = gh + (size_t)r * 768;
    float i  = sigf(gxr[m] + ghr[m]);
    float fx = gxr[256 + m];
    float f0 = sigf(fx + f0p[idx]);
    float f1 = sigf(fx + f1p[idx]);
    float u  = tanhf(gxr[512 + m] + ghr[256 + m]);
    float o  = sigf(gxr[768 + m] + ghr[512 + m]);
    size_t c0i = ((size_t)(2 * node + 1) * BATCH + b) * MDIM + m;
    size_t c1i = ((size_t)(2 * node + 2) * BATCH + b) * MDIM + m;
    float c = i * u + f0 * cbuf[c0i] + f1 * cbuf[c1i];
    float h = o * tanhf(c);
    size_t oi = ((size_t)node * BATCH + b) * MDIM + m;
    hbuf[oi] = h; cbuf[oi] = c;
}

// Tree cell, leaf (h_tilde = 0, no children): c = sig(i)*tanh(u_pre); h = sig(o)*tanh(c)
__global__ void final_gates_leaf(const float* __restrict__ gx,
                                 float* __restrict__ hbuf, float* __restrict__ cbuf,
                                 int base, int rows)
{
    long long idx = (long long)blockIdx.x * blockDim.x + threadIdx.x;
    if (idx >= (long long)rows * MDIM) return;
    int r = (int)(idx / MDIM), m = (int)(idx % MDIM);
    int p = r >> 7, b = r & 127;
    const float* gxr = gx + (size_t)r * 1024;
    float c = sigf(gxr[m]) * tanhf(gxr[512 + m]);
    float h = sigf(gxr[768 + m]) * tanhf(c);
    size_t oi = ((size_t)(base + p) * BATCH + b) * MDIM + m;
    hbuf[oi] = h; cbuf[oi] = c;
}

// dst[(colOff+n)*dstLd + k] = (bf16) src[k*Nc + n]   (transpose + convert)
__global__ void transpose_w(bf16_t* __restrict__ dst, const float* __restrict__ src,
                            int K, int Nc, int colOff, int dstLd)
{
    long long idx = (long long)blockIdx.x * blockDim.x + threadIdx.x;
    if (idx >= (long long)K * Nc) return;
    int k = (int)(idx / Nc), n = (int)(idx % Nc);
    dst[(size_t)(colOff + n) * dstLd + k] = (bf16_t)src[(size_t)k * Nc + n];
}

__global__ void vec_add(float* __restrict__ d, const float* __restrict__ a,
                        const float* __restrict__ b, int n)
{
    int i = blockIdx.x * blockDim.x + threadIdx.x;
    if (i < n) d[i] = a[i] + b[i];
}

__global__ void vec_copy(float* __restrict__ d, const float* __restrict__ s, int n)
{
    int i = blockIdx.x * blockDim.x + threadIdx.x;
    if (i < n) d[i] = s[i];
}

__global__ void copy_out(float* __restrict__ out, const float* __restrict__ hbuf, int n)
{
    int i = blockIdx.x * blockDim.x + threadIdx.x;
    if (i < n) out[i] = hbuf[i];   // node 0 block == first B*M floats (node-major)
}

// ---------------------------------------------------------------------------
// Host orchestration
// ---------------------------------------------------------------------------
static inline int nblk(long long n) { return (int)((n + 255) / 256); }

static void launch_gemm(const bf16_t* A, int lda, const bf16_t* Bt, int ldb,
                        const float* bias, const float* Cin, float* C, int ldc,
                        int rows, int cols, int K, hipStream_t s)
{
    dim3 g(cols / 64, rows / 128), b(128);
    gemm_bf16_wmma<<<g, b, 0, s>>>(A, lda, Bt, ldb, bias, Cin, C, ldc, K);
}

extern "C" void kernel_launch(void* const* d_in, const int* in_sizes, int n_in,
                              void* d_out, int out_size, void* d_ws, size_t ws_size,
                              hipStream_t stream)
{
    (void)in_sizes; (void)n_in; (void)out_size; (void)ws_size;

    const float* wte     = (const float*)d_in[0];
    const float* rel     = (const float*)d_in[1];
    // d_in[2] = n_nodes (compile-time 511)
    const float* Wix = (const float*)d_in[3];  const float* bix  = (const float*)d_in[4];
    const float* Wfx = (const float*)d_in[5];  const float* bfx  = (const float*)d_in[6];
    const float* Wux = (const float*)d_in[7];  const float* bux  = (const float*)d_in[8];
    const float* Wox = (const float*)d_in[9];  const float* box_ = (const float*)d_in[10];
    const float* Wih = (const float*)d_in[11]; const float* bih  = (const float*)d_in[12];
    const float* Wfh = (const float*)d_in[13]; const float* bfh  = (const float*)d_in[14];
    const float* Wuh = (const float*)d_in[15]; const float* buh  = (const float*)d_in[16];
    const float* Woh = (const float*)d_in[17]; const float* boh  = (const float*)d_in[18];
    const float* Wseq_ih = (const float*)d_in[19];
    const float* Wseq_hh = (const float*)d_in[20];
    const float* bseq_ih = (const float*)d_in[21];
    const float* bseq_hh = (const float*)d_in[22];

    // ---- workspace carving (256B aligned) ----
    size_t off = 0;
    char* base_ws = (char*)d_ws;
    auto carve = [&](size_t bytes) -> char* {
        char* p = base_ws + off;
        off += (bytes + 255) & ~(size_t)255;
        return p;
    };
    const size_t BM = (size_t)BATCH * MDIM;
    float*  hbuf  = (float*)carve((size_t)NNODES * BM * 4);
    float*  cbuf  = (float*)carve((size_t)NNODES * BM * 4);
    float*  bufG  = (float*)carve((size_t)32768 * 1024 * 4);  // g0 / gx
    float*  bufG2 = (float*)carve((size_t)16384 * 1024 * 4);  // g1 / gh
    bf16_t* Xbf   = (bf16_t*)carve((size_t)32768 * 512 * 2);
    bf16_t* A0    = (bf16_t*)carve((size_t)16384 * 320 * 2);
    bf16_t* A1    = (bf16_t*)carve((size_t)16384 * 320 * 2);
    float*  hs    = (float*)carve((size_t)16384 * 256 * 4);
    float*  cs    = (float*)carve((size_t)16384 * 256 * 4);
    bf16_t* hsb   = (bf16_t*)carve((size_t)16384 * 256 * 2);
    float*  f0p   = (float*)carve((size_t)16384 * 256 * 4);
    float*  f1p   = (float*)carve((size_t)16384 * 256 * 4);
    bf16_t* Wseq_ih_t = (bf16_t*)carve((size_t)1024 * 320 * 2);
    bf16_t* Wseq_hh_t = (bf16_t*)carve((size_t)1024 * 256 * 2);
    bf16_t* Wx4_t     = (bf16_t*)carve((size_t)1024 * 512 * 2);  // [Wix|Wfx|Wux|Wox]
    bf16_t* Wh3_t     = (bf16_t*)carve((size_t)768 * 256 * 2);   // [Wih|Wuh|Woh]
    bf16_t* Wfh_t     = (bf16_t*)carve((size_t)256 * 256 * 2);
    float*  bseq = (float*)carve(1024 * 4);
    float*  bx4  = (float*)carve(1024 * 4);
    float*  bh3  = (float*)carve(768 * 4);

    // ---- weight prep (bf16, column-major) ----
    transpose_w<<<nblk(320LL * 1024), 256, 0, stream>>>(Wseq_ih_t, Wseq_ih, 320, 1024, 0, 320);
    transpose_w<<<nblk(256LL * 1024), 256, 0, stream>>>(Wseq_hh_t, Wseq_hh, 256, 1024, 0, 256);
    transpose_w<<<nblk(512LL * 256), 256, 0, stream>>>(Wx4_t, Wix, 512, 256, 0,   512);
    transpose_w<<<nblk(512LL * 256), 256, 0, stream>>>(Wx4_t, Wfx, 512, 256, 256, 512);
    transpose_w<<<nblk(512LL * 256), 256, 0, stream>>>(Wx4_t, Wux, 512, 256, 512, 512);
    transpose_w<<<nblk(512LL * 256), 256, 0, stream>>>(Wx4_t, Wox, 512, 256, 768, 512);
    transpose_w<<<nblk(256LL * 256), 256, 0, stream>>>(Wh3_t, Wih, 256, 256, 0,   256);
    transpose_w<<<nblk(256LL * 256), 256, 0, stream>>>(Wh3_t, Wuh, 256, 256, 256, 256);
    transpose_w<<<nblk(256LL * 256), 256, 0, stream>>>(Wh3_t, Woh, 256, 256, 512, 256);
    transpose_w<<<nblk(256LL * 256), 256, 0, stream>>>(Wfh_t, Wfh, 256, 256, 0,   256);
    vec_add<<<4, 256, 0, stream>>>(bseq, bseq_ih, bseq_hh, 1024);
    vec_copy<<<1, 256, 0, stream>>>(bx4 + 0,   bix,  256);
    vec_copy<<<1, 256, 0, stream>>>(bx4 + 256, bfx,  256);
    vec_copy<<<1, 256, 0, stream>>>(bx4 + 512, bux,  256);
    vec_copy<<<1, 256, 0, stream>>>(bx4 + 768, box_, 256);
    vec_copy<<<1, 256, 0, stream>>>(bh3 + 0,   bih, 256);
    vec_copy<<<1, 256, 0, stream>>>(bh3 + 256, buh, 256);
    vec_copy<<<1, 256, 0, stream>>>(bh3 + 512, boh, 256);

    // ---- level-by-level bottom-up (9 sequential phases) ----
    for (int L = 8; L >= 0; --L) {
        int baseN = (1 << L) - 1;
        int P     = 1 << L;
        int rows  = P * BATCH;

        pack_x<<<nblk((long long)rows * INDIM), 256, 0, stream>>>(Xbf, wte, baseN, rows);

        if (L == 8) {
            // leaves: h_tilde = 0, no children
            launch_gemm(Xbf, 512, Wx4_t, 512, bx4, nullptr, bufG, 1024, rows, 1024, 512, stream);
            final_gates_leaf<<<nblk((long long)rows * MDIM), 256, 0, stream>>>(
                bufG, hbuf, cbuf, baseN, rows);
        } else {
            pack_child<<<nblk((long long)rows * 320), 256, 0, stream>>>(A0, hbuf, rel, baseN, 1, rows);
            pack_child<<<nblk((long long)rows * 320), 256, 0, stream>>>(A1, hbuf, rel, baseN, 2, rows);

            // seq-LSTM over the two children (sequential dependency)
            launch_gemm(A0, 320, Wseq_ih_t, 320, bseq, nullptr, bufG, 1024, rows, 1024, 320, stream);
            seq_cell0<<<nblk((long long)rows * MDIM), 256, 0, stream>>>(bufG, hs, cs, hsb, rows);
            launch_gemm(A1, 320, Wseq_ih_t, 320, bseq, nullptr, bufG2, 1024, rows, 1024, 320, stream);
            launch_gemm(hsb, 256, Wseq_hh_t, 256, nullptr, bufG2, bufG2, 1024, rows, 1024, 256, stream);
            seq_cell1<<<nblk((long long)rows * MDIM), 256, 0, stream>>>(bufG2, hs, cs, hsb, rows);

            // tree gates
            launch_gemm(Xbf, 512, Wx4_t, 512, bx4, nullptr, bufG, 1024, rows, 1024, 512, stream);
            launch_gemm(hsb, 256, Wh3_t, 256, bh3, nullptr, bufG2, 768, rows, 768, 256, stream);
            launch_gemm(A0, 320, Wfh_t, 256, bfh, nullptr, f0p, 256, rows, 256, 256, stream); // h_c0 = A0[:, :256]
            launch_gemm(A1, 320, Wfh_t, 256, bfh, nullptr, f1p, 256, rows, 256, 256, stream);

            final_gates<<<nblk((long long)rows * MDIM), 256, 0, stream>>>(
                bufG, bufG2, f0p, f1p, hbuf, cbuf, baseN, rows);
        }
    }

    copy_out<<<nblk((long long)BATCH * MDIM), 256, 0, stream>>>((float*)d_out, hbuf, BATCH * MDIM);
}
